// Dcls2d_12713103196285
// MI455X (gfx1250) — compile-verified, matrix operand order checked by reading, not run on hardware
//
#include <hip/hip_runtime.h>
#include <hip/hip_bf16.h>

// ---------------------------------------------------------------------------
// DCLS-2D (dilated conv with learnable spacings), MI455X / gfx1250, wave32.
// Dense 9x9 kernel built once, pre-swizzled into WMMA A-fragment order as
// split bf16 (hi/lo); implicit-GEMM conv with v_wmma_f32_16x16x32_bf16,
// bf16x3 compensation (~fp32 accuracy).
// Round 4: two output rows per block sharing A fragments (6 WMMAs per A-tile
// load pair), 3-slot LDS ring of fragment-ordered halo rows, one barrier per
// kh step.
// ---------------------------------------------------------------------------

typedef __bf16 bf16_t;
typedef __attribute__((ext_vector_type(2)))  __bf16 v2bf;
typedef __attribute__((ext_vector_type(16))) __bf16 v16bf;
typedef __attribute__((ext_vector_type(8)))  float  v8f;

#define CIN   128
#define COUT  128
#define BATCH 16
#define HH    112
#define WW    112
#define KHW   3
#define DIL   4
#define DD    9            // dilated kernel extent
#define PADP  4
#define NPOS  81           // 9*9 taps
#define NCHUNK 4           // 128 cin / 32
#define HALO_C 24          // staged columns: 16 px + 8 halo
#define COL_H  40          // halfwords per column slot (32 data + 8 pad = 80B)
#define ROW_H  (NCHUNK * HALO_C * COL_H)           // 3840 hw per plane/slot
#define NSLOT  3                                   // halo-row ring slots
#define FRAG_ELEMS (8 * NPOS * NCHUNK * 32 * 16)   // 1,327,104 per array
#define KDENSE_ELEMS (COUT * CIN * NPOS)           // 1,327,104 floats

// ---------------------------------------------------------------------------
__global__ __launch_bounds__(256) void zero_f32(float* p, int n) {
    int i = blockIdx.x * 256 + threadIdx.x;
    if (i < n) p[i] = 0.0f;
}

// Bilinear scatter of 3x3 weights into the dense 9x9 kernel.
__global__ __launch_bounds__(256) void build_dense_kernel(
    const float* __restrict__ weight,   // (O, C, 3, 3)
    const float* __restrict__ P,        // (2, O, C, 3, 3)
    float* __restrict__ Kd)             // (O, C, 81)
{
    int idx = blockIdx.x * 256 + threadIdx.x;
    const int NT = COUT * CIN * KHW * KHW;
    if (idx >= NT) return;
    int kw = idx % KHW;
    int kh = (idx / KHW) % KHW;
    int c  = (idx / (KHW * KHW)) % CIN;
    int o  = idx / (KHW * KHW * CIN);

    int widx = ((o * CIN + c) * KHW + kh) * KHW + kw;
    float w  = weight[widx];
    float p0 = P[widx];
    float p1 = P[COUT * CIN * KHW * KHW + widx];

    float ph = fminf(fmaxf((float)(kh * DIL) + p0, 0.0f), (float)(DD - 1));
    float pw = fminf(fmaxf((float)(kw * DIL) + p1, 0.0f), (float)(DD - 1));
    float fh = floorf(ph), fw = floorf(pw);
    float rh = ph - fh,    rw = pw - fw;
    int ih  = (int)fh,     iw  = (int)fw;
    int ih1 = (ih + 1 < DD) ? ih + 1 : DD - 1;
    int iw1 = (iw + 1 < DD) ? iw + 1 : DD - 1;

    float* base = Kd + (size_t)(o * CIN + c) * NPOS;
    atomicAdd(base + ih  * DD + iw , w * (1.0f - rh) * (1.0f - rw));
    atomicAdd(base + ih1 * DD + iw , w * rh          * (1.0f - rw));
    atomicAdd(base + ih  * DD + iw1, w * (1.0f - rh) * rw);
    atomicAdd(base + ih1 * DD + iw1, w * rh          * rw);
}

// k-index inside the 16-bit 16x32 fragment (CDNA5 ISA 7.12.2).
__device__ __forceinline__ int frag_k(int lane, int i) {
    return i + ((i >= 8) ? 8 : 0) + ((lane >= 16) ? 8 : 0);
}
// fragment-order position of cin-within-chunk k (inverse of frag_k);
// even k pairs stay adjacent.
__device__ __forceinline__ int frag_idx(int k) {
    return (k < 8) ? k : (k < 16) ? k + 8 : (k < 24) ? k - 8 : k;
}

// Re-swizzle dense K into A-fragment order, split into bf16 hi/lo.
// Layout: [cout_tile(8)][pos(81)][chunk(4)][lane(32)][halfword(16)]
__global__ __launch_bounds__(256) void build_a_frags(
    const float* __restrict__ Kd,
    bf16_t* __restrict__ A_hi,
    bf16_t* __restrict__ A_lo)
{
    int idx = blockIdx.x * 256 + threadIdx.x;
    if (idx >= FRAG_ELEMS) return;
    int i    = idx & 15;
    int t    = idx >> 4;
    int lane = t & 31;  t >>= 5;
    int chunk= t & 3;   t >>= 2;
    int pos  = t % NPOS;
    int ct   = t / NPOS;

    int cout = ct * 16 + (lane & 15);
    int cin  = chunk * 32 + frag_k(lane, i);

    float v = Kd[(size_t)(cout * CIN + cin) * NPOS + pos];
    bf16_t hi = (bf16_t)v;
    A_hi[idx] = hi;
    A_lo[idx] = (bf16_t)(v - (float)hi);
}

// ---------------------------------------------------------------------------
// Stage one halo row (absolute x row hh, cols w0-4 .. w0+19, all 128 cin)
// into LDS in fragment k-order, split bf16 hi/lo. 1536 even-k pairs / 256
// threads = 6 iterations; each: 2 coalesced global b32 loads -> 1 packed
// ds_store_b32 per plane.
__device__ __forceinline__ void stage_row(
    const float* __restrict__ x,
    bf16_t* __restrict__ dHhi, bf16_t* __restrict__ dHlo,
    int b, int hh, int w0, int tid)
{
    int hhc = (hh < 0) ? 0 : ((hh > HH - 1) ? HH - 1 : hh);
    bool rowok = (hh == hhc);

    #pragma unroll
    for (int j = 0; j < 6; ++j) {
        int p    = tid + 256 * j;        // 0..1535
        int col  = p % HALO_C;
        int kp2  = p / HALO_C;           // 0..63
        int cin0 = kp2 * 2;
        int k    = cin0 & 31;
        int chunk= cin0 >> 5;
        int idx  = frag_idx(k);          // even -> pair (idx, idx+1)

        int ww2 = w0 + col - PADP;
        int wwc = (ww2 < 0) ? 0 : ((ww2 > WW - 1) ? WW - 1 : ww2);
        bool ok = rowok & (ww2 == wwc);

        const float* px = x + (size_t)((b * CIN + cin0) * HH + hhc) * WW + wwc;
        float v0 = px[0];
        float v1 = px[(size_t)HH * WW];
        v0 = ok ? v0 : 0.0f;
        v1 = ok ? v1 : 0.0f;
        bf16_t h0 = (bf16_t)v0, h1 = (bf16_t)v1;
        v2bf hp, lp;
        hp[0] = h0;                       hp[1] = h1;
        lp[0] = (bf16_t)(v0 - (float)h0); lp[1] = (bf16_t)(v1 - (float)h1);

        int eh = chunk * (HALO_C * COL_H) + col * COL_H + idx;
        *(v2bf*)(dHhi + eh) = hp;
        *(v2bf*)(dHlo + eh) = lp;
    }
}

// ---------------------------------------------------------------------------
// Main implicit-GEMM conv. Block = (b, output rows h0/h0+1, 16-px strip);
// 8 waves cover 128 Cout. Per kh step: both rows share the A fragments
// (6 WMMAs per A hi/lo pair); halo rows live in a 3-slot LDS ring, one
// barrier per step. 216 WMMAs per wave per step.
__global__ __launch_bounds__(256) void dcls_conv(
    const float* __restrict__ x,        // (B, Cin, H, W)
    const float* __restrict__ bias,     // (Cout,)
    const bf16_t* __restrict__ A_hi,
    const bf16_t* __restrict__ A_lo,
    float* __restrict__ out)            // (B, Cout, H, W)
{
    __shared__ __align__(16) bf16_t sHhi[NSLOT][ROW_H];
    __shared__ __align__(16) bf16_t sHlo[NSLOT][ROW_H];

    const int WTILES = WW / 16;                 // 7
    const int HPAIRS = HH / 2;                  // 56
    int blk = blockIdx.x;
    int wt  = blk % WTILES;
    int hr  = (blk / WTILES) % HPAIRS;
    int b   = blk / (WTILES * HPAIRS);
    int h0  = hr * 2;                           // output rows h0, h0+1
    int w0  = wt * 16;

    int tid  = threadIdx.x;
    int wave = tid >> 5;          // cout tile 0..7
    int lane = tid & 31;
    int n    = lane & 15;
    int grp  = lane >> 4;
    int bofs = n * COL_H + grp * 16;   // lane's halfword offset in a column band

    v8f acc0 = {};
    v8f acc1 = {};

    // prologue: stage halo rows r=0 (h0-4) and r=1 (h0-3) into slots 0,1
    stage_row(x, sHhi[0], sHlo[0], b, h0 - PADP,     w0, tid);
    stage_row(x, sHhi[1], sHlo[1], b, h0 - PADP + 1, w0, tid);
    __syncthreads();

    for (int kh = 0; kh < DD; ++kh) {
        int s0 = kh % NSLOT;            // halo row for output row h0
        int s1 = (kh + 1) % NSLOT;      // halo row for output row h0+1
        const bf16_t* H0hi = sHhi[s0];
        const bf16_t* H0lo = sHlo[s0];
        const bf16_t* H1hi = sHhi[s1];
        const bf16_t* H1lo = sHlo[s1];

        for (int kw = 0; kw < DD; ++kw) {
            int pos = kh * DD + kw;

            // prefetch next tap's A tiles (global_prefetch_b8)
            if (pos + 1 < NPOS) {
                size_t pf = ((((size_t)wave * NPOS + (pos + 1)) * NCHUNK) * 32 + lane) * 16;
                __builtin_prefetch((const void*)(A_hi + pf), 0, 3);
                __builtin_prefetch((const void*)(A_lo + pf), 0, 3);
            }

            size_t abase = (((size_t)wave * NPOS + pos) * NCHUNK) * 512 + lane * 16;
            int    bband = kw * COL_H + bofs;
            #pragma unroll
            for (int chunk = 0; chunk < NCHUNK; ++chunk) {
                v16bf a_hi = *(const v16bf*)(A_hi + abase + chunk * 512);
                v16bf a_lo = *(const v16bf*)(A_lo + abase + chunk * 512);
                int coff = chunk * (HALO_C * COL_H) + bband;
                v16bf b0_hi = *(const v16bf*)(H0hi + coff);
                v16bf b0_lo = *(const v16bf*)(H0lo + coff);
                v16bf b1_hi = *(const v16bf*)(H1hi + coff);
                v16bf b1_lo = *(const v16bf*)(H1lo + coff);
                acc0 = __builtin_amdgcn_wmma_f32_16x16x32_bf16(
                           false, a_hi, false, b0_hi, (short)0, acc0, false, false);
                acc1 = __builtin_amdgcn_wmma_f32_16x16x32_bf16(
                           false, a_hi, false, b1_hi, (short)0, acc1, false, false);
                acc0 = __builtin_amdgcn_wmma_f32_16x16x32_bf16(
                           false, a_hi, false, b0_lo, (short)0, acc0, false, false);
                acc1 = __builtin_amdgcn_wmma_f32_16x16x32_bf16(
                           false, a_hi, false, b1_lo, (short)0, acc1, false, false);
                acc0 = __builtin_amdgcn_wmma_f32_16x16x32_bf16(
                           false, a_lo, false, b0_hi, (short)0, acc0, false, false);
                acc1 = __builtin_amdgcn_wmma_f32_16x16x32_bf16(
                           false, a_lo, false, b1_hi, (short)0, acc1, false, false);
            }
        }

        // stage halo row kh+2 into the slot last read at step kh-1
        if (kh + 2 <= DD) {
            int sn = (kh + 2) % NSLOT;
            stage_row(x, sHhi[sn], sHlo[sn], b, h0 - PADP + kh + 2, w0, tid);
        }
        __syncthreads();
    }

    // ---- store: C/D 16x16 f32 layout (lane<16: M=r, N=lane; lane>=16: M=r+8)
    int mofs = grp * 8;
    int co0  = wave * 16;
    #pragma unroll
    for (int r = 0; r < 8; ++r) {
        int co = co0 + r + mofs;
        float bs = bias[co];
        size_t o0 = ((size_t)(b * COUT + co) * HH + h0) * WW + w0 + n;
        out[o0]      = acc0[r] + bs;
        out[o0 + WW] = acc1[r] + bs;
    }
}

// ---------------------------------------------------------------------------
extern "C" void kernel_launch(void* const* d_in, const int* in_sizes, int n_in,
                              void* d_out, int out_size, void* d_ws, size_t ws_size,
                              hipStream_t stream) {
    (void)in_sizes; (void)n_in; (void)out_size; (void)ws_size;
    const float* x      = (const float*)d_in[0];
    const float* weight = (const float*)d_in[1];
    const float* P      = (const float*)d_in[2];
    const float* bias   = (const float*)d_in[3];
    float*       out    = (float*)d_out;

    // workspace carve-up (~10.6 MB total)
    float*  Kd   = (float*)d_ws;
    bf16_t* A_hi = (bf16_t*)((char*)d_ws + (size_t)KDENSE_ELEMS * 4);
    bf16_t* A_lo = A_hi + FRAG_ELEMS;

    zero_f32<<<(KDENSE_ELEMS + 255) / 256, 256, 0, stream>>>(Kd, KDENSE_ELEMS);

    const int NSCAT = COUT * CIN * KHW * KHW;                     // 147,456
    build_dense_kernel<<<(NSCAT + 255) / 256, 256, 0, stream>>>(weight, P, Kd);

    build_a_frags<<<(FRAG_ELEMS + 255) / 256, 256, 0, stream>>>(Kd, A_hi, A_lo);

    const int NBLK = BATCH * (HH / 2) * (WW / 16);                // 6,272
    dcls_conv<<<NBLK, 256, 0, stream>>>(x, bias, A_hi, A_lo, out);
}
